// GAT__20117626814612
// MI455X (gfx1250) — compile-verified, matrix operand order checked by reading, not run on hardware
//
#include <hip/hip_runtime.h>
#include <hip/hip_bf16.h>
#include <math.h>

// ---------------- types ----------------
typedef __attribute__((ext_vector_type(16))) __bf16 bf16x16;
typedef __attribute__((ext_vector_type(8)))  __bf16 bf16x8;
typedef __attribute__((ext_vector_type(8)))  float  v8f;

__device__ __forceinline__ __bf16 to_bf16(float f) {
    unsigned u = __builtin_bit_cast(unsigned, f);
    unsigned r = u + 0x7FFFu + ((u >> 16) & 1u);   // round-to-nearest-even
    unsigned short h = (unsigned short)(r >> 16);
    return __builtin_bit_cast(__bf16, h);
}

__device__ __forceinline__ float leaky02(float x) { return x > 0.f ? x : 0.2f * x; }
__device__ __forceinline__ float eluf(float x)    { return x > 0.f ? x : expm1f(x); }

// ---------------- small utility kernels ----------------
__global__ void zero_i32(int* __restrict__ p, int n) {
    int i = blockIdx.x * blockDim.x + threadIdx.x;
    if (i < n) p[i] = 0;
}

__global__ void hist_dst(const int* __restrict__ dst, int E, int* __restrict__ cnt) {
    int i = blockIdx.x * blockDim.x + threadIdx.x;
    if (i < E) atomicAdd(&cnt[dst[i]], 1);
}

// single-block Hillis-Steele scan over tiles of 1024 (exclusive -> rowptr)
__global__ void scan_counts(const int* __restrict__ cnt, int* __restrict__ rowptr, int n) {
    __shared__ int lds[1024];
    __shared__ int carry;
    int t = threadIdx.x;
    if (t == 0) carry = 0;
    for (int base = 0; base < n; base += 1024) {
        __syncthreads();
        int v = (base + t < n) ? cnt[base + t] : 0;
        lds[t] = v;
        __syncthreads();
        for (int off = 1; off < 1024; off <<= 1) {
            int x = (t >= off) ? lds[t - off] : 0;
            __syncthreads();
            lds[t] += x;
            __syncthreads();
        }
        int incl = lds[t];
        int c = carry;
        if (base + t < n) rowptr[base + t] = c + incl - v;
        __syncthreads();
        if (t == 1023) carry = c + lds[1023];
    }
    __syncthreads();
    if (t == 0) rowptr[n] = carry;
}

__global__ void csr_fill(const int* __restrict__ dst, int E,
                         const int* __restrict__ rowptr, int* __restrict__ fill,
                         int* __restrict__ order) {
    int e = blockIdx.x * blockDim.x + threadIdx.x;
    if (e < E) {
        int d = dst[e];
        int pos = rowptr[d] + atomicAdd(&fill[d], 1);
        order[pos] = e;
    }
}

__global__ void cvt_bf16(const float* __restrict__ in, __bf16* __restrict__ out, int n) {
    int i = blockIdx.x * blockDim.x + threadIdx.x;
    if (i < n) out[i] = to_bf16(in[i]);
}

// in: [R,C] row-major f32 -> out: [C,R] bf16 (transposed)
__global__ void transpose_bf16(const float* __restrict__ in, __bf16* __restrict__ out,
                               int R, int C) {
    int i = blockIdx.x * blockDim.x + threadIdx.x;
    if (i < R * C) {
        int r = i / C, c = i % C;
        out[(size_t)c * R + r] = to_bf16(in[i]);
    }
}

// ---------------- WMMA GEMM: C[M,Nout] = A[M,K] * B[K,Nout], B given as Bt[Nout,K] ----------------
// One wave computes 16 rows x (16*NT) cols; A fragment reused across NT n-tiles.
template <int NT>
__global__ void gemm_bf16_wmma(const __bf16* __restrict__ A,
                               const __bf16* __restrict__ Bt,
                               float* __restrict__ C,
                               int M, int K, int Nout) {
    int wave = threadIdx.x >> 5;
    int lane = threadIdx.x & 31;
    int mBlk = blockIdx.x * 16;
    int nBase = wave * (16 * NT);
    if (nBase >= Nout) return;

    int mRow = lane & 15;
    if (mBlk + mRow >= M) mRow = 0;          // clamp loads for ragged tail
    int kHalf = (lane >> 4) * 8;             // lanes 0-15: K 0..7/16..23 ; lanes 16-31: +8
    int nCol  = lane & 15;

    const __bf16* pa = A + (size_t)(mBlk + mRow) * K + kHalf;

    v8f z = {0.f, 0.f, 0.f, 0.f, 0.f, 0.f, 0.f, 0.f};
    v8f acc[NT];
#pragma unroll
    for (int t = 0; t < NT; ++t) acc[t] = z;

    for (int k0 = 0; k0 < K; k0 += 32) {
        bf16x16 a;
        bf16x8 alo = *(const bf16x8*)(pa + k0);
        bf16x8 ahi = *(const bf16x8*)(pa + k0 + 16);
#pragma unroll
        for (int i = 0; i < 8; ++i) { a[i] = alo[i]; a[i + 8] = ahi[i]; }
#pragma unroll
        for (int t = 0; t < NT; ++t) {
            const __bf16* pb = Bt + (size_t)(nBase + t * 16 + nCol) * K + kHalf + k0;
            bf16x8 blo = *(const bf16x8*)pb;
            bf16x8 bhi = *(const bf16x8*)(pb + 16);
            bf16x16 b;
#pragma unroll
            for (int i = 0; i < 8; ++i) { b[i] = blo[i]; b[i + 8] = bhi[i]; }
            acc[t] = __builtin_amdgcn_wmma_f32_16x16x32_bf16(
                /*neg_a=*/false, a, /*neg_b=*/false, b,
                /*c_mod=*/(short)0, acc[t], /*reuse_a=*/false, /*reuse_b=*/false);
        }
    }

    int mTop = (lane >> 4) * 8;              // lanes 0-15 hold M 0..7, lanes 16-31 hold M 8..15
#pragma unroll
    for (int t = 0; t < NT; ++t) {
#pragma unroll
        for (int r = 0; r < 8; ++r) {
            int mr = mBlk + mTop + r;
            if (mr < M) C[(size_t)mr * Nout + nBase + t * 16 + nCol] = acc[t][r];
        }
    }
}

// ---------------- attention logits: el/er[n,h] = sum_d f[n,h,d]*al/ar[h,d] (D=32) ----------------
__global__ void attn_logits(const float* __restrict__ f,
                            const float* __restrict__ al, const float* __restrict__ ar,
                            float* __restrict__ el, float* __restrict__ er,
                            int Nn, int H) {
    int i = blockIdx.x * blockDim.x + threadIdx.x;
    if (i >= Nn * H) return;
    int node = i / H, h = i % H;
    const float4* fp = (const float4*)(f + ((size_t)node * H + h) * 32);
    const float4* ap = (const float4*)(al + h * 32);
    const float4* rp = (const float4*)(ar + h * 32);
    float sl = 0.f, sr = 0.f;
#pragma unroll
    for (int j = 0; j < 8; ++j) {
        float4 fv = fp[j], av = ap[j], rv = rp[j];
        sl += fv.x * av.x + fv.y * av.y + fv.z * av.z + fv.w * av.w;
        sr += fv.x * rv.x + fv.y * rv.y + fv.z * rv.z + fv.w * rv.w;
    }
    el[i] = sl; er[i] = sr;
}

// ---------------- layer-1 aggregation: H=8, D=32, wave per dst node, no atomics ----------------
// lane l: head h = l>>2, dims (l&3)*8 .. +7  (flat channel index = l*8 .. l*8+7)
__global__ void gat_agg_h8(const float* __restrict__ f,   // [Nn,256]
                           const float* __restrict__ el,  // [Nn,8]
                           const float* __restrict__ er,  // [Nn,8]
                           const int* __restrict__ rowptr,
                           const int* __restrict__ order,
                           const int* __restrict__ srcv,
                           const float* __restrict__ bias, // [256]
                           float* __restrict__ out,        // [Nn,256] (ELU applied)
                           int Nn) {
    int wave = threadIdx.x >> 5, lane = threadIdx.x & 31;
    int v = blockIdx.x * (blockDim.x >> 5) + wave;
    if (v >= Nn) return;
    int h = lane >> 2, slot = lane & 3;
    int beg = rowptr[v], end = rowptr[v + 1];
    float erv = er[(size_t)v * 8 + h];

    // pass 1: online softmax max/denom, 4 lanes per head each striding edges
    float m = -__builtin_inff(), s = 0.f;
    for (int i = beg + slot; i < end; i += 4) {
        int sn = srcv[order[i]];
        float xv = leaky02(el[(size_t)sn * 8 + h] + erv);
        if (xv > m) { s = s * __expf(m - xv) + 1.f; m = xv; }
        else        { s += __expf(xv - m); }
    }
#pragma unroll
    for (int off = 1; off <= 2; off <<= 1) {
        float mo = __shfl_xor(m, off, 32);
        float so = __shfl_xor(s, off, 32);
        float mn = fmaxf(m, mo);
        float sa = (m  > -1e37f) ? s  * __expf(m  - mn) : 0.f;
        float sb = (mo > -1e37f) ? so * __expf(mo - mn) : 0.f;
        m = mn; s = sa + sb;
    }
    float inv = (s > 0.f) ? 1.f / s : 0.f;

    // pass 2: weighted accumulation of gathered rows (coalesced 2x float4 per lane)
    float a0=0,a1=0,a2=0,a3=0,a4=0,a5=0,a6=0,a7=0;
    for (int i = beg; i < end; ++i) {
        int sn = srcv[order[i]];
        float xv = leaky02(el[(size_t)sn * 8 + h] + erv);
        float alpha = __expf(xv - m) * inv;
        const float4* fr = (const float4*)(f + (size_t)sn * 256 + lane * 8);
        float4 p = fr[0], q = fr[1];
        a0 += p.x * alpha; a1 += p.y * alpha; a2 += p.z * alpha; a3 += p.w * alpha;
        a4 += q.x * alpha; a5 += q.y * alpha; a6 += q.z * alpha; a7 += q.w * alpha;
    }
    const float4* bp = (const float4*)(bias + lane * 8);
    float4 b0 = bp[0], b1 = bp[1];
    float4 o0, o1;
    o0.x = eluf(a0 + b0.x); o0.y = eluf(a1 + b0.y); o0.z = eluf(a2 + b0.z); o0.w = eluf(a3 + b0.w);
    o1.x = eluf(a4 + b1.x); o1.y = eluf(a5 + b1.y); o1.z = eluf(a6 + b1.z); o1.w = eluf(a7 + b1.w);
    float4* op = (float4*)(out + (size_t)v * 256 + lane * 8);
    op[0] = o0; op[1] = o1;
}

// ---------------- layer-2 aggregation: H=1, D=32, wave per dst node ----------------
__global__ void gat_agg_h1(const float* __restrict__ f,   // [Nn,32]
                           const float* __restrict__ el,  // [Nn]
                           const float* __restrict__ er,  // [Nn]
                           const int* __restrict__ rowptr,
                           const int* __restrict__ order,
                           const int* __restrict__ srcv,
                           const float* __restrict__ bias, // [32]
                           float* __restrict__ out,        // [Nn,32]
                           int Nn) {
    int wave = threadIdx.x >> 5, lane = threadIdx.x & 31;
    int v = blockIdx.x * (blockDim.x >> 5) + wave;
    if (v >= Nn) return;
    int beg = rowptr[v], end = rowptr[v + 1];
    float erv = er[v];

    float m = -__builtin_inff(), s = 0.f;
    for (int i = beg + lane; i < end; i += 32) {
        int sn = srcv[order[i]];
        float xv = leaky02(el[sn] + erv);
        if (xv > m) { s = s * __expf(m - xv) + 1.f; m = xv; }
        else        { s += __expf(xv - m); }
    }
#pragma unroll
    for (int off = 1; off <= 16; off <<= 1) {
        float mo = __shfl_xor(m, off, 32);
        float so = __shfl_xor(s, off, 32);
        float mn = fmaxf(m, mo);
        float sa = (m  > -1e37f) ? s  * __expf(m  - mn) : 0.f;
        float sb = (mo > -1e37f) ? so * __expf(mo - mn) : 0.f;
        m = mn; s = sa + sb;
    }
    float inv = (s > 0.f) ? 1.f / s : 0.f;

    float acc = 0.f;
    for (int i = beg; i < end; ++i) {
        int sn = srcv[order[i]];
        float xv = leaky02(el[sn] + erv);
        float alpha = __expf(xv - m) * inv;
        acc += f[(size_t)sn * 32 + lane] * alpha;   // coalesced across the wave
    }
    out[(size_t)v * 32 + lane] = acc + bias[lane];  // H=1 -> mean over heads is identity
}

// ---------------- launcher ----------------
extern "C" void kernel_launch(void* const* d_in, const int* in_sizes, int n_in,
                              void* d_out, int out_size, void* d_ws, size_t ws_size,
                              hipStream_t stream) {
    const float* x   = (const float*)d_in[0];
    const int*   src = (const int*)  d_in[1];
    const int*   dst = (const int*)  d_in[2];
    const float* W1  = (const float*)d_in[3];
    const float* al1 = (const float*)d_in[4];
    const float* ar1 = (const float*)d_in[5];
    const float* b1  = (const float*)d_in[6];
    const float* W2  = (const float*)d_in[7];
    const float* al2 = (const float*)d_in[8];
    const float* ar2 = (const float*)d_in[9];
    const float* b2  = (const float*)d_in[10];
    (void)n_in; (void)out_size; (void)ws_size;

    const int F = 128, C1 = 256, H1 = 8, C2 = 32;
    const int Nn = in_sizes[0] / F;
    const int E  = in_sizes[1];
    float* outp = (float*)d_out;

    char* base = (char*)d_ws;
    size_t off = 0;
    auto alloc = [&](size_t bytes) -> void* {
        void* p = base + off;
        off = (off + bytes + 255) & ~(size_t)255;
        return p;
    };
    __bf16* xb   = (__bf16*)alloc((size_t)Nn * F  * 2);
    __bf16* W1t  = (__bf16*)alloc((size_t)C1 * F  * 2);   // [256,128]
    __bf16* W2t  = (__bf16*)alloc((size_t)C2 * C1 * 2);   // [32,256]
    float*  f1   = (float*) alloc((size_t)Nn * C1 * 4);
    float*  el1  = (float*) alloc((size_t)Nn * H1 * 4);
    float*  er1  = (float*) alloc((size_t)Nn * H1 * 4);
    float*  h1   = (float*) alloc((size_t)Nn * C1 * 4);
    __bf16* h1b  = (__bf16*)alloc((size_t)Nn * C1 * 2);
    float*  f2   = (float*) alloc((size_t)Nn * C2 * 4);
    float*  el2  = (float*) alloc((size_t)Nn * 4);
    float*  er2  = (float*) alloc((size_t)Nn * 4);
    int*    cnt  = (int*)   alloc((size_t)Nn * 4);
    int*    rowp = (int*)   alloc((size_t)(Nn + 1) * 4);
    int*    ordr = (int*)   alloc((size_t)E * 4);

    const int TB = 256;
    auto nb = [TB](long n) { return (unsigned)((n + TB - 1) / TB); };

    // ---- CSR by destination (replaces ~230M float atomics with ~1.6M int atomics) ----
    zero_i32  <<<nb(Nn), TB, 0, stream>>>(cnt, Nn);
    hist_dst  <<<nb(E),  TB, 0, stream>>>(dst, E, cnt);
    scan_counts<<<1, 1024, 0, stream>>>(cnt, rowp, Nn);
    zero_i32  <<<nb(Nn), TB, 0, stream>>>(cnt, Nn);
    csr_fill  <<<nb(E),  TB, 0, stream>>>(dst, E, rowp, cnt, ordr);

    // ---- precision conversion for the matrix pipes ----
    cvt_bf16      <<<nb((long)Nn * F), TB, 0, stream>>>(x,  xb,  Nn * F);
    transpose_bf16<<<nb((long)F * C1), TB, 0, stream>>>(W1, W1t, F,  C1);
    transpose_bf16<<<nb((long)C1 * C2),TB, 0, stream>>>(W2, W2t, C1, C2);

    // ---- layer 1 ----
    gemm_bf16_wmma<4><<<dim3((Nn + 15) / 16), 128, 0, stream>>>(xb, W1t, f1, Nn, F, C1);
    attn_logits<<<nb((long)Nn * H1), TB, 0, stream>>>(f1, al1, ar1, el1, er1, Nn, H1);
    gat_agg_h8 <<<(Nn + 7) / 8, 256, 0, stream>>>(f1, el1, er1, rowp, ordr, src, b1, h1, Nn);

    // ---- layer 2 ----
    cvt_bf16<<<nb((long)Nn * C1), TB, 0, stream>>>(h1, h1b, Nn * C1);
    gemm_bf16_wmma<2><<<dim3((Nn + 15) / 16), 32, 0, stream>>>(h1b, W2t, f2, Nn, C1, C2);
    attn_logits<<<nb(Nn), TB, 0, stream>>>(f2, al2, ar2, el2, er2, Nn, 1);
    gat_agg_h1 <<<(Nn + 7) / 8, 256, 0, stream>>>(f2, el2, er2, rowp, ordr, src, b2, outp, Nn);
}
